// Codebook_8916352107068
// MI455X (gfx1250) — compile-verified, matrix operand order checked by reading, not run on hardware
//
#include <hip/hip_runtime.h>
#include <hip/hip_bf16.h>
#include <math.h>

typedef __attribute__((ext_vector_type(16))) _Float16 v16h;
typedef __attribute__((ext_vector_type(8)))  float    v8f;
typedef __attribute__((ext_vector_type(4)))  int      v4i;

#define BDIM   8
#define TDIM   1024
#define DIN    1024
#define NCODES 1024
#define NCB    8
#define MROWS  (BDIM * TDIM)    // 8192
#define NCOLS  (NCB * NCODES)   // 8192
#define MTILE  32               // rows per block (2 WMMA row-tiles)

#if __has_builtin(__builtin_amdgcn_global_load_async_to_lds_b128)
#define HAVE_ASYNC_LDS 1
#else
#define HAVE_ASYNC_LDS 0
#endif

// ---------------------------------------------------------------------------
// Kernel A: fp32 -> fp16 precast (vectorized x4)
// ---------------------------------------------------------------------------
__global__ void cast_f32_f16(const float* __restrict__ in,
                             _Float16* __restrict__ out, int n4) {
  int i = blockIdx.x * blockDim.x + threadIdx.x;
  if (i < n4) {
    float4 v = ((const float4*)in)[i];
    union { _Float16 h[4]; uint2 u; } p;
    p.h[0] = (_Float16)v.x; p.h[1] = (_Float16)v.y;
    p.h[2] = (_Float16)v.z; p.h[3] = (_Float16)v.w;
    ((uint2*)out)[i] = p.u;
  }
}

// ---------------------------------------------------------------------------
// Kernel B: fused GEMM (x @ proj_w^T) + online logsumexp cross-entropy.
// Block = 256 threads (8 waves), M-tile = 32 rows. Each wave computes
// 2 M-tiles x 4 N-tiles = 8 accumulators: one A-fragment pair feeds 8 WMMAs,
// halving B-matrix L2 traffic vs a 16-row block. Double-buffered K loop.
// xp (256 MB) is never materialized.
// ---------------------------------------------------------------------------
__global__ __launch_bounds__(256) void ce_gemm_kernel(
    const _Float16* __restrict__ Xh, const _Float16* __restrict__ Wh,
    const float* __restrict__ proj_b, const int* __restrict__ target,
    float* __restrict__ loss_out) {
  __shared__ _Float16 Ash[MTILE * DIN];       // 64 KB A tile
  __shared__ float stage_max[8][MTILE];
  __shared__ float stage_sum[8][MTILE];
  __shared__ float acc_max[MTILE][NCB];
  __shared__ float acc_sum[MTILE][NCB];
  __shared__ float tgt_logit[MTILE][NCB];
  __shared__ int   tgt_s[MTILE][NCB];

  const int tid    = threadIdx.x;
  const int wave   = tid >> 5;
  const int lane   = tid & 31;
  const int lane15 = lane & 15;
  const int khalf  = (lane >> 4) << 3;        // 0 or 8 (A/B fragment K split)
  const int rhi    = (lane >> 4) << 3;        // C fragment row offset
  const int m0     = blockIdx.x * MTILE;

  // ---- Stage A tile (32 rows x 1024 f16 = 64 KB, contiguous) into LDS ----
  {
    const uint4* src = (const uint4*)(Xh + (size_t)m0 * DIN);
    uint4* dst = (uint4*)Ash;
#if HAVE_ASYNC_LDS
    for (int i = tid; i < (MTILE * DIN) / 8; i += 256) {
      __builtin_amdgcn_global_load_async_to_lds_b128(
          (__attribute__((address_space(1))) v4i*)(src + i),
          (__attribute__((address_space(3))) v4i*)(dst + i),
          /*offset=*/0, /*cpol=*/0);
    }
#else
    for (int i = tid; i < (MTILE * DIN) / 8; i += 256) dst[i] = src[i];
#endif
  }
  {
    int r = tid >> 3, k = tid & 7;            // 256 threads = 32 rows x 8 cb
    tgt_s[r][k]     = target[(m0 + r) * NCB + k];
    acc_max[r][k]   = -1e30f;
    acc_sum[r][k]   = 0.f;
    tgt_logit[r][k] = 0.f;
  }
#if HAVE_ASYNC_LDS
#if __has_builtin(__builtin_amdgcn_s_wait_asynccnt)
  __builtin_amdgcn_s_wait_asynccnt(0);
#else
  asm volatile("s_wait_asynccnt 0x0" ::: "memory");
#endif
#endif
  __syncthreads();

  const _Float16* arow[2];
  arow[0] = Ash + lane15 * DIN;
  arow[1] = Ash + (16 + lane15) * DIN;

  // 16 chunks of 512 columns; each wave owns 4 consecutive 16-col tiles.
  for (int ci = 0; ci < NCOLS / 512; ++ci) {
    const int kb       = ci >> 1;             // codebook (1024 cols each)
    const int tilebase = ci * 512 + wave * 64;

    const _Float16* brow[4];
#pragma unroll
    for (int t = 0; t < 4; ++t)
      brow[t] = Wh + (size_t)(tilebase + 16 * t + lane15) * DIN;

    v8f c[2][4];
#pragma unroll
    for (int mt = 0; mt < 2; ++mt)
#pragma unroll
      for (int t = 0; t < 4; ++t) c[mt][t] = (v8f){};

    // ---- K loop, double-buffered fragments ----
    v16h a[2][2], b[2][4];
    {
      const int kk = khalf;
#pragma unroll
      for (int mt = 0; mt < 2; ++mt) {
        ((uint4*)&a[0][mt])[0] = *(const uint4*)(arow[mt] + kk);
        ((uint4*)&a[0][mt])[1] = *(const uint4*)(arow[mt] + kk + 16);
      }
#pragma unroll
      for (int t = 0; t < 4; ++t) {
        ((uint4*)&b[0][t])[0] = *(const uint4*)(brow[t] + kk);
        ((uint4*)&b[0][t])[1] = *(const uint4*)(brow[t] + kk + 16);
      }
    }
#pragma unroll 2
    for (int kki = 0; kki < DIN / 32; ++kki) {
      const int cur = kki & 1, nxt = cur ^ 1;
      if (kki + 1 < DIN / 32) {
        const int kk = (kki + 1) * 32 + khalf;
#pragma unroll
        for (int mt = 0; mt < 2; ++mt) {
          ((uint4*)&a[nxt][mt])[0] = *(const uint4*)(arow[mt] + kk);
          ((uint4*)&a[nxt][mt])[1] = *(const uint4*)(arow[mt] + kk + 16);
        }
#pragma unroll
        for (int t = 0; t < 4; ++t) {
          ((uint4*)&b[nxt][t])[0] = *(const uint4*)(brow[t] + kk);
          ((uint4*)&b[nxt][t])[1] = *(const uint4*)(brow[t] + kk + 16);
        }
      }
#pragma unroll
      for (int mt = 0; mt < 2; ++mt)
#pragma unroll
        for (int t = 0; t < 4; ++t) {
          c[mt][t] = __builtin_amdgcn_wmma_f32_16x16x32_f16(
              false, a[cur][mt], false, b[cur][t], (short)0, c[mt][t],
              false, false);
        }
    }

    // ---- Epilogue: bias, target logit, per-row logsumexp over 64 cols ----
    float M[2][8], S[2][8];
#pragma unroll
    for (int mt = 0; mt < 2; ++mt)
#pragma unroll
      for (int r = 0; r < 8; ++r) { M[mt][r] = -1e30f; S[mt][r] = 0.f; }

#pragma unroll
    for (int t = 0; t < 4; ++t) {
      const int col    = tilebase + 16 * t + lane15;
      const float bias = proj_b[col];
      const int cincb  = col - (kb << 10);    // column within this codebook
#pragma unroll
      for (int mt = 0; mt < 2; ++mt) {
#pragma unroll
        for (int r = 0; r < 8; ++r) {
          float v = c[mt][t][r] + bias;
          int row = 16 * mt + r + rhi;
          if (tgt_s[row][kb] == cincb) tgt_logit[row][kb] = v;
          float m = v;
          m = fmaxf(m, __shfl_xor(m, 1, 32));
          m = fmaxf(m, __shfl_xor(m, 2, 32));
          m = fmaxf(m, __shfl_xor(m, 4, 32));
          m = fmaxf(m, __shfl_xor(m, 8, 32));
          float s = __expf(v - m);
          s += __shfl_xor(s, 1, 32);
          s += __shfl_xor(s, 2, 32);
          s += __shfl_xor(s, 4, 32);
          s += __shfl_xor(s, 8, 32);
          float nm = fmaxf(M[mt][r], m);
          S[mt][r] = S[mt][r] * __expf(M[mt][r] - nm) + s * __expf(m - nm);
          M[mt][r] = nm;
        }
      }
    }
    if (lane15 == 0) {
#pragma unroll
      for (int mt = 0; mt < 2; ++mt)
#pragma unroll
        for (int r = 0; r < 8; ++r) {
          stage_max[wave][16 * mt + r + rhi] = M[mt][r];
          stage_sum[wave][16 * mt + r + rhi] = S[mt][r];
        }
    }
    __syncthreads();
    if (tid < MTILE) {                        // one thread per row merges 8 waves
      float am = acc_max[tid][kb], as = acc_sum[tid][kb];
      for (int w = 0; w < 8; ++w) {
        float m = stage_max[w][tid], s = stage_sum[w][tid];
        float nm = fmaxf(am, m);
        as = as * __expf(am - nm) + s * __expf(m - nm);
        am = nm;
      }
      acc_max[tid][kb] = am; acc_sum[tid][kb] = as;
    }
    __syncthreads();
  }

  {
    int r = tid >> 3, k = tid & 7;            // 256 threads = 32 rows x 8 cb
    loss_out[(size_t)(m0 + r) * NCB + k] =
        acc_max[r][k] + __logf(acc_sum[r][k]) - tgt_logit[r][k];
  }
}

// ---------------------------------------------------------------------------
// Kernel C: xw = softmax(x @ wproj^T) over 8 codebooks, then
// emb = sum_k xw[k] * codebook[target[k], k, :]. One block per (b,t) row.
// ---------------------------------------------------------------------------
__global__ __launch_bounds__(256) void xw_emb_kernel(
    const float* __restrict__ x, const int* __restrict__ target,
    const float* __restrict__ codebook, const float* __restrict__ wproj,
    float* __restrict__ emb) {
  __shared__ float wsum[8][NCB];
  __shared__ float wgt[NCB];
  __shared__ int   tgt[NCB];
  const int bt   = blockIdx.x;
  const int tid  = threadIdx.x;
  const int lane = tid & 31, wave = tid >> 5;
  const int d0   = tid * 4;

  float4 xv = *(const float4*)(x + (size_t)bt * DIN + d0);
#pragma unroll
  for (int k = 0; k < NCB; ++k) {
    float4 wv = *(const float4*)(wproj + (size_t)k * DIN + d0);
    float s = xv.x * wv.x + xv.y * wv.y + xv.z * wv.z + xv.w * wv.w;
    s += __shfl_xor(s, 16, 32);
    s += __shfl_xor(s, 8, 32);
    s += __shfl_xor(s, 4, 32);
    s += __shfl_xor(s, 2, 32);
    s += __shfl_xor(s, 1, 32);
    if (lane == 0) wsum[wave][k] = s;
  }
  if (tid < NCB) tgt[tid] = target[(size_t)bt * NCB + tid];
  __syncthreads();

  if (tid == 0) {
    float lg[NCB], mx = -1e30f;
    for (int k = 0; k < NCB; ++k) {
      float s = 0.f;
      for (int w = 0; w < 8; ++w) s += wsum[w][k];
      lg[k] = s; mx = fmaxf(mx, s);
    }
    float den = 0.f;
    for (int k = 0; k < NCB; ++k) { lg[k] = __expf(lg[k] - mx); den += lg[k]; }
    float inv = 1.f / den;
    for (int k = 0; k < NCB; ++k) wgt[k] = lg[k] * inv;
  }
  __syncthreads();

  float4 acc = {0.f, 0.f, 0.f, 0.f};
#pragma unroll
  for (int k = 0; k < NCB; ++k) {
    const float4 cv =
        *(const float4*)(codebook + ((size_t)tgt[k] * NCB + k) * DIN + d0);
    float w = wgt[k];
    acc.x += w * cv.x; acc.y += w * cv.y; acc.z += w * cv.z; acc.w += w * cv.w;
  }
  *(float4*)(emb + (size_t)bt * DIN + d0) = acc;
}

// ---------------------------------------------------------------------------
extern "C" void kernel_launch(void* const* d_in, const int* in_sizes, int n_in,
                              void* d_out, int out_size, void* d_ws, size_t ws_size,
                              hipStream_t stream) {
  const float* x        = (const float*)d_in[0];
  const int*   target   = (const int*)d_in[1];
  const float* codebook = (const float*)d_in[2];
  const float* proj_w   = (const float*)d_in[3];
  const float* proj_b   = (const float*)d_in[4];
  const float* wproj_w  = (const float*)d_in[5];

  float* emb  = (float*)d_out;                       // (8,1024,1024)
  float* loss = (float*)d_out + (size_t)MROWS * DIN; // (8,1024,8)

  _Float16* Xh = (_Float16*)d_ws;                        // 16 MB
  _Float16* Wh = (_Float16*)d_ws + (size_t)MROWS * DIN;  // 16 MB

  const int n4 = (MROWS * DIN) / 4;  // 2,097,152 float4s each
  cast_f32_f16<<<(n4 + 255) / 256, 256, 0, stream>>>(x, Xh, n4);
  cast_f32_f16<<<(n4 + 255) / 256, 256, 0, stream>>>(proj_w, Wh, n4);
  ce_gemm_kernel<<<MROWS / MTILE, 256, 0, stream>>>(Xh, Wh, proj_b, target, loss);
  xw_emb_kernel<<<MROWS, 256, 0, stream>>>(x, target, codebook, wproj_w, emb);
}
